// CTCA_61280593379575
// MI455X (gfx1250) — compile-verified, matrix operand order checked by reading, not run on hardware
//
#include <hip/hip_runtime.h>

// ---------------------------------------------------------------------------
// Nystromformer attention for MI455X (gfx1250).
// All GEMMs on v_wmma_f32_16x16x32_f16 (f32 -> f16 staged in LDS, f32 acc).
// All LDS operands stored contiguous-K so every fragment load is ds_load_b128.
// attn3 @ v fused flash-style (2 chained WMMAs) to avoid a 256MB tensor.
// ---------------------------------------------------------------------------

typedef __attribute__((ext_vector_type(16))) _Float16 v16h;
typedef __attribute__((ext_vector_type(4)))  _Float16 v4h;
typedef __attribute__((ext_vector_type(8)))  float    v8f;
typedef __attribute__((ext_vector_type(4)))  float    v4f;

#define B_SZ   2
#define HEADS  8
#define SEQ_N  4096
#define NQ     1024
#define DIM    512
#define DHEAD  64
#define LM     256
#define INNER  512           // HEADS*DHEAD
#define BH     (B_SZ*HEADS)

// ---------------- WMMA helpers (layouts per CDNA5 ISA 7.12.2) ----------------

__device__ __forceinline__ v8f wmma_f16(v16h a, v16h b, v8f c) {
  return __builtin_amdgcn_wmma_f32_16x16x32_f16(false, a, false, b,
                                                (short)0, c, false, false);
}

// 16x32 fragment from a row-major LDS tile with contiguous K (ld % 8 == 0 so
// both 16-byte chunks are aligned -> two ds_load_b128 per fragment).
// lane<16: rows 0..15, K {0..7,16..23}; lane>=16: same rows, K {8..15,24..31}.
__device__ __forceinline__ v16h frag_k(const _Float16* s, int ld, int lane) {
  const _Float16* p = s + (lane & 15) * ld + ((lane & 16) ? 8 : 0);
  v16h f;
#pragma unroll
  for (int i = 0; i < 8; ++i) {
    f[i]     = p[i];
    f[i + 8] = p[16 + i];
  }
  return f;
}

// ---------------- Generic batched WMMA GEMM ----------------
// C = alpha * (A @ B[^T]) + diag * I + bias.
// 128 threads = 4 waves in a 2x2 grid; each wave computes (TM/2)x(TN/2) as
// (TM/32)x(TN/32) 16x16 WMMA tiles. K staged 64 at a time, % 64 == 0.
// Batch z = blockIdx.z -> (b = z/H, h = z%H) with per-operand strides.

template <int TM, int TN, bool TRANSB>
__global__ __launch_bounds__(128) void gemm_wmma(
    const float* __restrict__ A, const float* __restrict__ B,
    const float* __restrict__ bias, float* __restrict__ C,
    int M, int N, int K, int lda, int ldb, int ldc,
    long long sAb, long long sAh, long long sBb, long long sBh,
    long long sCb, long long sCh, int H, float alpha, float diag) {
  constexpr int TMT = TM / 32;      // 16-tiles per wave, rows
  constexpr int TNT = TN / 32;      // 16-tiles per wave, cols
  constexpr int LDK = 72;           // 64 K + pad; 144B rows keep 16B alignment
  constexpr int LTN = (TN == 128) ? 7 : 6;

  int z = blockIdx.z;
  int bb = z / H, hh = z % H;
  A += (long long)bb * sAb + (long long)hh * sAh;
  B += (long long)bb * sBb + (long long)hh * sBh;
  C += (long long)bb * sCb + (long long)hh * sCh;

  int row0 = blockIdx.y * TM;
  int col0 = blockIdx.x * TN;
  if (row0 >= M || col0 >= N) return;

  __shared__ _Float16 As[TM][LDK];  // [m][k]
  __shared__ _Float16 Bs[TN][LDK];  // [n][k]  (N-major -> contiguous-K frags)

  int tid = threadIdx.x;
  int wave = tid >> 5, lane = tid & 31;
  int wm = (wave >> 1) * (TM / 2);
  int wn = (wave & 1) * (TN / 2);

  v8f acc[TMT][TNT] = {};
  for (int k0 = 0; k0 < K; k0 += 64) {
    // Stage A: TM x 64, float4 global reads, 8B f16 LDS writes.
    for (int i = tid * 4; i < TM * 64; i += 128 * 4) {
      int r = i >> 6, c = i & 63;
      v4f v = *(const v4f*)&A[(long long)(row0 + r) * lda + k0 + c];
      v4h hv = {(_Float16)v.x, (_Float16)v.y, (_Float16)v.z, (_Float16)v.w};
      *(v4h*)&As[r][c] = hv;
    }
    // Stage B into [n][k].
    if (TRANSB) {  // B is (N x K) row-major: contiguous in k
      for (int i = tid * 4; i < TN * 64; i += 128 * 4) {
        int n = i >> 6, k = i & 63;
        v4f v = *(const v4f*)&B[(long long)(col0 + n) * ldb + k0 + k];
        v4h hv = {(_Float16)v.x, (_Float16)v.y, (_Float16)v.z, (_Float16)v.w};
        *(v4h*)&Bs[n][k] = hv;
      }
    } else {       // B is (K x N) row-major: contiguous in n, transposed store
      for (int i = tid * 4; i < TN * 64; i += 128 * 4) {
        int k = i >> LTN, n = i & (TN - 1);
        v4f v = *(const v4f*)&B[(long long)(k0 + k) * ldb + col0 + n];
        Bs[n + 0][k] = (_Float16)v.x;
        Bs[n + 1][k] = (_Float16)v.y;
        Bs[n + 2][k] = (_Float16)v.z;
        Bs[n + 3][k] = (_Float16)v.w;
      }
    }
    if (k0 + 64 < K) {  // prefetch next K panel (global_prefetch_b8)
      __builtin_prefetch(&A[(long long)(row0 + (tid & (TM - 1))) * lda + k0 + 64], 0, 1);
      if (TRANSB)
        __builtin_prefetch(&B[(long long)(col0 + (tid & (TN - 1))) * ldb + k0 + 64], 0, 1);
      else
        __builtin_prefetch(&B[(long long)(k0 + 64 + (tid & 31)) * ldb + col0], 0, 1);
    }
    __syncthreads();

#pragma unroll
    for (int ks = 0; ks < 64; ks += 32) {
      v16h af[TMT], bf[TNT];
#pragma unroll
      for (int ti = 0; ti < TMT; ++ti) af[ti] = frag_k(&As[wm + ti * 16][ks], LDK, lane);
#pragma unroll
      for (int tj = 0; tj < TNT; ++tj) bf[tj] = frag_k(&Bs[wn + tj * 16][ks], LDK, lane);
#pragma unroll
      for (int ti = 0; ti < TMT; ++ti)
#pragma unroll
        for (int tj = 0; tj < TNT; ++tj)
          acc[ti][tj] = wmma_f16(af[ti], bf[tj], acc[ti][tj]);
    }
    __syncthreads();
  }

  int crow = (lane & 16) ? 8 : 0;
  int ccol = lane & 15;
#pragma unroll
  for (int ti = 0; ti < TMT; ++ti)
#pragma unroll
    for (int tj = 0; tj < TNT; ++tj)
#pragma unroll
      for (int r = 0; r < 8; ++r) {
        int gr = row0 + wm + ti * 16 + crow + r;
        int gc = col0 + wn + tj * 16 + ccol;
        float v = alpha * acc[ti][tj][r];
        if (gr == gc) v += diag;
        if (bias) v += bias[gc];
        C[(long long)gr * ldc + gc] = v;
      }
}

// ---------------- Small helper kernels ----------------

__global__ void landmark_pool(const float* __restrict__ kv,
                              float* __restrict__ kland) {
  int idx = blockIdx.x * blockDim.x + threadIdx.x;
  if (idx >= BH * LM * DHEAD) return;
  int dd = idx & 63;
  int mm = (idx >> 6) & (LM - 1);
  int bh = idx >> 14;
  int b = bh >> 3, h = bh & 7;
  const float* base = kv + (long long)b * SEQ_N * (2 * INNER) + h * DHEAD + dd;
  float s = 0.f;
#pragma unroll 4
  for (int j = 0; j < 16; ++j)
    s += base[(long long)(mm * 16 + j) * (2 * INNER)];
  kland[idx] = s;
}

__global__ __launch_bounds__(256) void softmax_rows(float* __restrict__ X) {
  __shared__ float red[256];
  long long row = blockIdx.x;
  float v = X[row * LM + threadIdx.x];
  red[threadIdx.x] = v;
  __syncthreads();
  for (int o = 128; o; o >>= 1) {
    if (threadIdx.x < o)
      red[threadIdx.x] = fmaxf(red[threadIdx.x], red[threadIdx.x + o]);
    __syncthreads();
  }
  float m = red[0];
  __syncthreads();
  float e = __expf(v - m);
  red[threadIdx.x] = e;
  __syncthreads();
  for (int o = 128; o; o >>= 1) {
    if (threadIdx.x < o) red[threadIdx.x] += red[threadIdx.x + o];
    __syncthreads();
  }
  X[row * LM + threadIdx.x] = e / red[0];
}

// Pass 1 of fused attn3@v: per-row max and 1/sum(exp) over n=4096.
__global__ __launch_bounds__(256) void attn_row_stats(
    const float* __restrict__ q, const float* __restrict__ kv,
    float* __restrict__ rmax, float* __restrict__ rinv) {
  int bh = blockIdx.z, b = bh >> 3, h = bh & 7;
  int qi = blockIdx.x;
  __shared__ float qrow[DHEAD];
  __shared__ float red[256];
  const float* qp = q + (long long)b * NQ * INNER + (long long)qi * INNER + h * DHEAD;
  if (threadIdx.x < DHEAD) qrow[threadIdx.x] = qp[threadIdx.x];
  __syncthreads();

  const float* kbase = kv + (long long)b * SEQ_N * (2 * INNER) + h * DHEAD;
  float s[16];
  float lmax = -3.0e38f;
#pragma unroll
  for (int t = 0; t < 16; ++t) {
    const float* kp = kbase + (long long)(threadIdx.x + t * 256) * (2 * INNER);
    float dot = 0.f;
#pragma unroll
    for (int dd = 0; dd < DHEAD; ++dd) dot = __fmaf_rn(qrow[dd], kp[dd], dot);
    s[t] = dot;
    lmax = fmaxf(lmax, dot);
  }
  red[threadIdx.x] = lmax;
  __syncthreads();
  for (int o = 128; o; o >>= 1) {
    if (threadIdx.x < o)
      red[threadIdx.x] = fmaxf(red[threadIdx.x], red[threadIdx.x + o]);
    __syncthreads();
  }
  float m = red[0];
  __syncthreads();
  float ls = 0.f;
#pragma unroll
  for (int t = 0; t < 16; ++t) ls += __expf(s[t] - m);
  red[threadIdx.x] = ls;
  __syncthreads();
  for (int o = 128; o; o >>= 1) {
    if (threadIdx.x < o) red[threadIdx.x] += red[threadIdx.x + o];
    __syncthreads();
  }
  if (threadIdx.x == 0) {
    rmax[(long long)bh * NQ + qi] = m;
    rinv[(long long)bh * NQ + qi] = 1.0f / red[0];
  }
}

// Pass 2: out3 = softmax(q k^T) @ v, fused via two chained WMMAs per chunk.
__global__ __launch_bounds__(128) void attn_av(
    const float* __restrict__ q, const float* __restrict__ kv,
    const float* __restrict__ rmax, const float* __restrict__ rinv,
    float* __restrict__ out3) {
  int bh = blockIdx.z, b = bh >> 3, h = bh & 7;
  int r0 = blockIdx.x * 64;
  int tid = threadIdx.x, wave = tid >> 5, lane = tid & 31;
  int wm = (wave >> 1) * 32, wn = (wave & 1) * 32;
  int sc = (wave & 1) * 16;  // S-tile column quadrant for this wave

  __shared__ _Float16 Qs[64][72];   // [row][d]        (A for S)
  __shared__ _Float16 Ksn[32][72];  // [n][d], N-major (B for S)
  __shared__ _Float16 Ps[64][40];   // [row][n] exp(S) (A for P@V)
  __shared__ _Float16 Vsn[64][40];  // [d][n], N-major (B for P@V)
  __shared__ float mrow[64], rr[64];

  const float* qbase = q + (long long)b * NQ * INNER + h * DHEAD;
  for (int i = tid * 4; i < 64 * 64; i += 128 * 4) {
    int r = i >> 6, c = i & 63;
    v4f v = *(const v4f*)&qbase[(long long)(r0 + r) * INNER + c];
    v4h hv = {(_Float16)v.x, (_Float16)v.y, (_Float16)v.z, (_Float16)v.w};
    *(v4h*)&Qs[r][c] = hv;
  }
  if (tid < 64) {
    mrow[tid] = rmax[(long long)bh * NQ + r0 + tid];
    rr[tid]   = rinv[(long long)bh * NQ + r0 + tid];
  }
  __syncthreads();

  const float* kb = kv + (long long)b * SEQ_N * (2 * INNER) + h * DHEAD;
  const float* vb = kb + INNER;  // v is second half of kv columns

  v8f acc[2][2] = {};
  for (int n0 = 0; n0 < SEQ_N; n0 += 32) {
    // Ksn[n][d] : contiguous reads and writes
    for (int i = tid * 4; i < 32 * 64; i += 128 * 4) {
      int nn = i >> 6, kk = i & 63;
      v4f v = *(const v4f*)&kb[(long long)(n0 + nn) * (2 * INNER) + kk];
      v4h hv = {(_Float16)v.x, (_Float16)v.y, (_Float16)v.z, (_Float16)v.w};
      *(v4h*)&Ksn[nn][kk] = hv;
    }
    // Vsn[d][n] : contiguous reads, transposed stores
    for (int i = tid * 4; i < 32 * 64; i += 128 * 4) {
      int nn = i >> 6, dd = i & 63;
      v4f v = *(const v4f*)&vb[(long long)(n0 + nn) * (2 * INNER) + dd];
      Vsn[dd + 0][nn] = (_Float16)v.x;
      Vsn[dd + 1][nn] = (_Float16)v.y;
      Vsn[dd + 2][nn] = (_Float16)v.z;
      Vsn[dd + 3][nn] = (_Float16)v.w;
    }
    if (n0 + 32 < SEQ_N) {
      __builtin_prefetch(&kb[(long long)(n0 + 32 + (tid & 31)) * (2 * INNER)], 0, 1);
      __builtin_prefetch(&vb[(long long)(n0 + 32 + (tid & 31)) * (2 * INNER)], 0, 1);
    }
    __syncthreads();

    // S = Q(64x64) @ K^T(64x32); this wave: rows wm..wm+31, cols sc..sc+15
    v8f s0 = {}, s1 = {};
#pragma unroll
    for (int ks = 0; ks < 64; ks += 32) {
      v16h aq0 = frag_k(&Qs[wm][ks], 72, lane);
      v16h aq1 = frag_k(&Qs[wm + 16][ks], 72, lane);
      v16h bk  = frag_k(&Ksn[sc][ks], 72, lane);
      s0 = wmma_f16(aq0, bk, s0);
      s1 = wmma_f16(aq1, bk, s1);
    }
    int crow = (lane & 16) ? 8 : 0;
    int ccol = lane & 15;
#pragma unroll
    for (int r = 0; r < 8; ++r) {
      int rr0 = wm + crow + r;
      int rr1 = wm + 16 + crow + r;
      Ps[rr0][sc + ccol] = (_Float16)__expf(s0[r] - mrow[rr0]);
      Ps[rr1][sc + ccol] = (_Float16)__expf(s1[r] - mrow[rr1]);
    }
    __syncthreads();

    // acc += P(64x32) @ V(32x64); this wave: 32x32 quadrant (wm, wn)
    v16h pa0 = frag_k(&Ps[wm][0], 40, lane);
    v16h pa1 = frag_k(&Ps[wm + 16][0], 40, lane);
    v16h vb0 = frag_k(&Vsn[wn][0], 40, lane);
    v16h vb1 = frag_k(&Vsn[wn + 16][0], 40, lane);
    acc[0][0] = wmma_f16(pa0, vb0, acc[0][0]);
    acc[0][1] = wmma_f16(pa0, vb1, acc[0][1]);
    acc[1][0] = wmma_f16(pa1, vb0, acc[1][0]);
    acc[1][1] = wmma_f16(pa1, vb1, acc[1][1]);
    __syncthreads();
  }

  float* ob = out3 + (long long)bh * NQ * DHEAD;
  int crow = (lane & 16) ? 8 : 0;
  int ccol = lane & 15;
#pragma unroll
  for (int ti = 0; ti < 2; ++ti)
#pragma unroll
    for (int tj = 0; tj < 2; ++tj)
#pragma unroll
      for (int r = 0; r < 8; ++r) {
        int lr = wm + ti * 16 + crow + r;
        int gc = wn + tj * 16 + ccol;
        ob[(long long)(r0 + lr) * DHEAD + gc] = acc[ti][tj][r] * rr[lr];
      }
}

// T = c*I - X, batched (z = bh), n x n row-major contiguous per batch.
__global__ void diag_sub(const float* __restrict__ X, float* __restrict__ T,
                         int n, float c) {
  long long base = (long long)blockIdx.z * n * n;
  int idx = blockIdx.x * blockDim.x + threadIdx.x;
  if (idx >= n * n) return;
  int i = idx / n, j = idx - i * n;
  T[base + idx] = ((i == j) ? c : 0.0f) - X[base + idx];
}

__global__ void init_scal(float* s) {
  if (threadIdx.x < 2) s[threadIdx.x] = 0.0f;
}

// max over rows of sum_j |attn[row][j]|  (reference "col" term)
__global__ void rowabs_max(const float* __restrict__ attn, float* __restrict__ m) {
  long long row = blockIdx.x * blockDim.x + threadIdx.x;
  if (row >= (long long)BH * NQ) return;
  const float* p = attn + row * LM;
  float s = 0.f;
  for (int j = 0; j < LM; ++j) s += fabsf(p[j]);
  atomicMax((int*)m, __float_as_int(s));  // values > 0: int order == float order
}

// max over (bh, col) of sum_i |attn[i][col]|  (reference "row" term)
__global__ void colabs_max(const float* __restrict__ attn, float* __restrict__ m) {
  int idx = blockIdx.x * blockDim.x + threadIdx.x;
  if (idx >= BH * LM) return;
  int bh = idx >> 8, j = idx & (LM - 1);
  const float* p = attn + (long long)bh * NQ * LM + j;
  float s = 0.f;
  for (int i = 0; i < NQ; ++i) s += fabsf(p[(long long)i * LM]);
  atomicMax((int*)m, __float_as_int(s));
}

// Z0[bh][j][i] = attn[bh][i][j] / (scal[0]*scal[1])
__global__ void init_z(const float* __restrict__ attn,
                       const float* __restrict__ scal, float* __restrict__ Z) {
  long long idx = (long long)blockIdx.x * blockDim.x + threadIdx.x;
  if (idx >= (long long)BH * LM * NQ) return;
  int i = idx & (NQ - 1);
  int j = (int)((idx >> 10) & (LM - 1));
  long long bh = idx >> 18;
  float inv = 1.0f / (scal[0] * scal[1]);
  Z[idx] = attn[bh * NQ * LM + (long long)i * LM + j] * inv;
}

// (b,h,nq,64) -> (b,nq,h*64)
__global__ void rearrange_out(const float* __restrict__ outh,
                              float* __restrict__ o) {
  long long idx = (long long)blockIdx.x * blockDim.x + threadIdx.x;
  if (idx >= (long long)B_SZ * NQ * INNER) return;
  int c = idx & (INNER - 1);
  long long r = idx >> 9;
  int b = (int)(r >> 10), i = (int)(r & (NQ - 1));
  int h = c >> 6, dd = c & 63;
  o[idx] = outh[((long long)(b * HEADS + h) * NQ + i) * DHEAD + dd];
}

// ---------------------------------------------------------------------------

extern "C" void kernel_launch(void* const* d_in, const int* in_sizes, int n_in,
                              void* d_out, int out_size, void* d_ws,
                              size_t ws_size, hipStream_t stream) {
  const float* x       = (const float*)d_in[0];
  const float* q_input = (const float*)d_in[1];
  const float* W_kv    = (const float*)d_in[2];
  const float* W_q     = (const float*)d_in[3];
  const float* W_out   = (const float*)d_in[4];
  const float* b_out   = (const float*)d_in[5];
  float* out = (float*)d_out;

  char* ws = (char*)d_ws;
  auto alloc = [&](size_t bytes) -> float* {
    char* p = ws;
    ws += (bytes + 255) & ~(size_t)255;
    return (float*)p;
  };
  float* kv    = alloc((size_t)B_SZ * SEQ_N * 2 * INNER * 4);  // 32 MB
  float* qb    = alloc((size_t)B_SZ * NQ * INNER * 4);         //  4 MB
  float* kland = alloc((size_t)BH * LM * DHEAD * 4);
  float* attn1 = alloc((size_t)BH * NQ * LM * 4);              // 16 MB
  float* rmaxA = alloc((size_t)BH * NQ * 4);
  float* rinvA = alloc((size_t)BH * NQ * 4);
  float* out3  = alloc((size_t)BH * NQ * DHEAD * 4);
  float* Za    = alloc((size_t)BH * LM * NQ * 4);
  float* Zb    = alloc((size_t)BH * LM * NQ * 4);
  float* XZ    = alloc((size_t)BH * NQ * NQ * 4);              // 64 MB
  float* T1    = alloc((size_t)BH * NQ * NQ * 4);              // 64 MB
  float* T2    = alloc((size_t)BH * NQ * NQ * 4);              // 64 MB
  float* tmp   = alloc((size_t)BH * LM * DHEAD * 4);
  float* outh  = alloc((size_t)BH * NQ * DHEAD * 4);
  float* rearr = alloc((size_t)B_SZ * NQ * INNER * 4);
  float* scal  = alloc(256);

  const dim3 blk(128);
  const float scale = 0.125f;  // DHEAD^-0.5

  // kv = x @ W_kv : (8192 x 512) @ (512 x 1024)
  gemm_wmma<128, 128, false><<<dim3(8, 64, 1), blk, 0, stream>>>(
      x, W_kv, nullptr, kv, 8192, 1024, 512, 512, 1024, 1024,
      0, 0, 0, 0, 0, 0, 1, 1.0f, 0.0f);
  // q = (q_input @ W_q) * scale : (2048 x 512) @ (512 x 512)
  gemm_wmma<128, 128, false><<<dim3(4, 16, 1), blk, 0, stream>>>(
      q_input, W_q, nullptr, qb, 2048, 512, 512, 512, 512, 512,
      0, 0, 0, 0, 0, 0, 1, scale, 0.0f);

  landmark_pool<<<(BH * LM * DHEAD + 255) / 256, 256, 0, stream>>>(kv, kland);

  // sim1 = q_bh @ kland_bh^T, batched over (b,h); attn1 = softmax(sim1)
  gemm_wmma<128, 128, true><<<dim3(LM / 128, NQ / 128, BH), blk, 0, stream>>>(
      qb, kland, nullptr, attn1, NQ, LM, DHEAD, INNER, DHEAD, LM,
      (long long)NQ * INNER, DHEAD,
      (long long)HEADS * LM * DHEAD, (long long)LM * DHEAD,
      (long long)HEADS * NQ * LM, (long long)NQ * LM, HEADS, 1.0f, 0.0f);
  softmax_rows<<<BH * NQ, 256, 0, stream>>>(attn1);

  // out3 = softmax(q k^T) @ v  (fused, no attn3 materialization)
  attn_row_stats<<<dim3(NQ, 1, BH), 256, 0, stream>>>(qb, kv, rmaxA, rinvA);
  attn_av<<<dim3(NQ / 64, 1, BH), blk, 0, stream>>>(qb, kv, rmaxA, rinvA, out3);

  // Moore-Penrose pinv of attn1 (== attn2)
  init_scal<<<1, 32, 0, stream>>>(scal);
  rowabs_max<<<(BH * NQ + 255) / 256, 256, 0, stream>>>(attn1, scal + 0);
  colabs_max<<<(BH * LM + 255) / 256, 256, 0, stream>>>(attn1, scal + 1);
  init_z<<<(BH * LM * NQ + 255) / 256, 256, 0, stream>>>(attn1, scal, Za);

  long long sQ = (long long)NQ * NQ;   // 1024x1024 batch stride
  long long sZ = (long long)LM * NQ;   // 256x1024 batch stride
  long long sP = (long long)NQ * LM;   // 1024x256 batch stride
  float* Zc = Za;
  float* Zn = Zb;
  for (int it = 0; it < 6; ++it) {
    // XZ = P @ Z                          (1024 x 256 x 1024)
    gemm_wmma<128, 128, false><<<dim3(8, 8, BH), blk, 0, stream>>>(
        attn1, Zc, nullptr, XZ, NQ, NQ, LM, LM, NQ, NQ,
        0, sP, 0, sZ, 0, sQ, BH, 1.0f, 0.0f);
    // T1 = 7I - XZ
    diag_sub<<<dim3(NQ * NQ / 256, 1, BH), 256, 0, stream>>>(XZ, T1, NQ, 7.0f);
    // T2 = 15I - XZ @ T1                  (1024^3, epilogue-fused)
    gemm_wmma<128, 128, false><<<dim3(8, 8, BH), blk, 0, stream>>>(
        XZ, T1, nullptr, T2, NQ, NQ, NQ, NQ, NQ, NQ,
        0, sQ, 0, sQ, 0, sQ, BH, -1.0f, 15.0f);
    // T3 = 13I - XZ @ T2  (into T1)
    gemm_wmma<128, 128, false><<<dim3(8, 8, BH), blk, 0, stream>>>(
        XZ, T2, nullptr, T1, NQ, NQ, NQ, NQ, NQ, NQ,
        0, sQ, 0, sQ, 0, sQ, BH, -1.0f, 13.0f);
    // Z = 0.25 * Z @ T3                   (256 x 1024 x 1024)
    gemm_wmma<128, 128, false><<<dim3(8, 2, BH), blk, 0, stream>>>(
        Zc, T1, nullptr, Zn, LM, NQ, NQ, NQ, NQ, NQ,
        0, sZ, 0, sQ, 0, sZ, BH, 0.25f, 0.0f);
    float* t = Zc; Zc = Zn; Zn = t;
  }

  // tmp = Z @ out3 : (256 x 1024) @ (1024 x 64)
  gemm_wmma<64, 64, false><<<dim3(1, 4, BH), blk, 0, stream>>>(
      Zc, out3, nullptr, tmp, LM, DHEAD, NQ, NQ, DHEAD, DHEAD,
      0, sZ, 0, (long long)NQ * DHEAD, 0, (long long)LM * DHEAD, BH, 1.0f, 0.0f);
  // outh = attn1 @ tmp : (1024 x 256) @ (256 x 64)
  gemm_wmma<64, 64, false><<<dim3(1, 16, BH), blk, 0, stream>>>(
      attn1, tmp, nullptr, outh, NQ, DHEAD, LM, LM, DHEAD, DHEAD,
      0, sP, 0, (long long)LM * DHEAD, 0, (long long)NQ * DHEAD, BH, 1.0f, 0.0f);

  // merge heads + output projection with bias
  rearrange_out<<<(B_SZ * NQ * INNER + 255) / 256, 256, 0, stream>>>(outh, rearr);
  gemm_wmma<128, 128, false><<<dim3(4, 16, 1), blk, 0, stream>>>(
      rearr, W_out, b_out, out, B_SZ * NQ, DIM, INNER, INNER, DIM, DIM,
      0, 0, 0, 0, 0, 0, 1, 1.0f, 0.0f);
}
